// message_pass_32615981646068
// MI455X (gfx1250) — matrix operand order repaired and checked by reading, not compiled, not run on hardware
//
#include <hip/hip_runtime.h>

// MI455X / gfx1250: per-sensor message passing block.
//   state1 = bmm(x, W1^T) + b1          (S=64 GEMMs, M=1024, N=512, K=512)
//   state2 = sum_s(state1) - state1
//   out    = bmm(state2, W3^T) + b3
// bf16x3 decomposition (hi*hi + lo*hi + hi*lo, fp32 accum) on
// V_WMMA_F32_16X16X32_BF16. Weights pre-split to bf16 hi/lo in ws and staged
// to LDS via GLOBAL_LOAD_ASYNC_TO_LDS_B128; activations split on the fly with
// v_perm-packed truncation and staged cooperatively through LDS.

#define B_DIM 1024
#define S_DIM 64
#define H_DIM 512

#define LDS_STRIDE 20  // dwords per 32-K row (16 data + 4 pad): 16B aligned, bank-spread

typedef __attribute__((ext_vector_type(16))) __bf16 v16bf;
typedef __attribute__((ext_vector_type(8)))  float  v8f;

union Frag {
    v16bf v;
    unsigned int u[8];
    uint4 q[2];
};

// dst = {bf16(f_odd), bf16(f_even)} via byte-perm of the two fp32 words.
__device__ __forceinline__ unsigned pack_hi_pair(unsigned u0, unsigned u1) {
    return __builtin_amdgcn_perm(u1, u0, 0x07060302u);
}

__device__ __forceinline__ void async_copy_b128(unsigned lds_off, const void* gptr) {
    unsigned long long ga = (unsigned long long)(uintptr_t)gptr;
    asm volatile("global_load_async_to_lds_b128 %0, %1, off"
                 :: "v"(lds_off), "v"(ga)
                 : "memory");
}
__device__ __forceinline__ void wait_asynccnt0() {
    asm volatile("s_wait_asynccnt 0" ::: "memory");
}

// Split fp32 -> separate bf16-hi / bf16-lo arrays (2 elems per u32), truncation split.
__global__ __launch_bounds__(256) void split_bf16x2_kernel(
    const float* __restrict__ src, unsigned int* __restrict__ hi2,
    unsigned int* __restrict__ lo2, int n2) {
    int i = blockIdx.x * 256 + threadIdx.x;
    if (i >= n2) return;
    float2 f = ((const float2*)src)[i];
    unsigned u0 = __float_as_uint(f.x), u1 = __float_as_uint(f.y);
    float h0 = __uint_as_float(u0 & 0xFFFF0000u);
    float h1 = __uint_as_float(u1 & 0xFFFF0000u);
    hi2[i] = pack_hi_pair(u0, u1);
    lo2[i] = pack_hi_pair(__float_as_uint(f.x - h0), __float_as_uint(f.y - h1));
}

// sums[b,h] = sum_s state1[b,s,h]
__global__ __launch_bounds__(256) void reduce_sum_kernel(
    const float* __restrict__ st1, float* __restrict__ sums) {
    size_t idx = (size_t)blockIdx.x * 256 + threadIdx.x;  // over B*H
    int b = (int)(idx / H_DIM);
    int h = (int)(idx % H_DIM);
    const float* p = st1 + (size_t)b * S_DIM * H_DIM + h;
    float a = 0.f;
#pragma unroll 8
    for (int s = 0; s < S_DIM; ++s) a += p[(size_t)s * H_DIM];
    sums[idx] = a;
}

// Per-sensor GEMM: out[m,s,n] = sum_k A[m,s,k]*W[s,n,k] + bias[s,n]
// MODE 0: A = A0 (= x).  MODE 1: A = Asum[m,k] - A0[m,s,k] (= state2).
// WG = 256 threads = 8 waves; WG tile 128Mx128N; wave tile 32Mx64N.
template <int MODE>
__global__ __launch_bounds__(256) void sensor_gemm_kernel(
    const float* __restrict__ A0,          // [B,S,H]
    const float* __restrict__ Asum,        // [B,H] (MODE1 only)
    const unsigned int* __restrict__ Whi,  // [S,H,H] bf16 hi, 2/u32
    const unsigned int* __restrict__ Wlo,  // [S,H,H] bf16 lo, 2/u32
    const float* __restrict__ bias,        // [S,H]
    float* __restrict__ out)               // [B,S,H]
{
    __shared__ unsigned lds_ahi[128 * LDS_STRIDE];
    __shared__ unsigned lds_alo[128 * LDS_STRIDE];
    __shared__ unsigned lds_bhi[128 * LDS_STRIDE];
    __shared__ unsigned lds_blo[128 * LDS_STRIDE];

    const int s    = blockIdx.z;
    const int mblk = blockIdx.y;
    const int nblk = blockIdx.x;
    const int tid  = threadIdx.x;
    const int wave = tid >> 5;
    const int lane = tid & 31;
    const int half = lane >> 4;  // 0: lanes 0-15, 1: lanes 16-31
    const int l15  = lane & 15;
    const int wm   = wave & 3;   // 4 waves along M
    const int wn   = wave >> 2;  // 2 waves along N
    const int mbase = mblk * 128 + wm * 32;
    const int nbase = nblk * 128 + wn * 64;

    // ---- async B-tile copy setup: 512 x 16B chunks per K-step over 256 thr ----
    const unsigned* gb_hi[2];
    const unsigned* gb_lo[2];
    unsigned ldsb_hi[2], ldsb_lo[2];
#pragma unroll
    for (int c = 0; c < 2; ++c) {
        int chunk = tid + c * 256;  // 0..511
        int col = chunk >> 2;       // 0..127 (local column)
        int sub = chunk & 3;        // which 4-dword piece of the 16-dword K row
        size_t rowoff = (((size_t)s * H_DIM + nblk * 128 + col) * H_DIM) >> 1;
        gb_hi[c] = Whi + rowoff + sub * 4;
        gb_lo[c] = Wlo + rowoff + sub * 4;
        ldsb_hi[c] = (unsigned)(uintptr_t)&lds_bhi[col * LDS_STRIDE + sub * 4];
        ldsb_lo[c] = (unsigned)(uintptr_t)&lds_blo[col * LDS_STRIDE + sub * 4];
    }

    // ---- cooperative A-tile loader: thread -> (row, 16-K half) ----
    const int lrow = tid >> 1;  // 0..127
    const int lkh  = tid & 1;   // 0..1
    const float* ld_arow =
        A0 + (((size_t)(mblk * 128 + lrow)) * S_DIM + s) * H_DIM + 16 * lkh;
    const float* ld_srow =
        (MODE == 1) ? (Asum + ((size_t)(mblk * 128 + lrow)) * H_DIM + 16 * lkh)
                    : ld_arow;
    unsigned* lda_hi = &lds_ahi[lrow * LDS_STRIDE + 8 * lkh];
    unsigned* lda_lo = &lds_alo[lrow * LDS_STRIDE + 8 * lkh];

    v8f zero = {};
    v8f acc[2][4];
#pragma unroll
    for (int i = 0; i < 2; ++i)
#pragma unroll
        for (int j = 0; j < 4; ++j) acc[i][j] = zero;

    for (int kk = 0; kk < H_DIM; kk += 32) {
        __syncthreads();  // previous tile fully consumed

        // 1) kick off async DMA of the bf16 weight tile into LDS
        async_copy_b128(ldsb_hi[0], gb_hi[0] + (kk >> 1));
        async_copy_b128(ldsb_hi[1], gb_hi[1] + (kk >> 1));
        async_copy_b128(ldsb_lo[0], gb_lo[0] + (kk >> 1));
        async_copy_b128(ldsb_lo[1], gb_lo[1] + (kk >> 1));

        // 2) load 16 fp32 activations, split hi/lo, stage to LDS
        float f[16];
        {
            const float4* pa = (const float4*)(ld_arow + kk);
            *(float4*)(f + 0)  = pa[0];
            *(float4*)(f + 4)  = pa[1];
            *(float4*)(f + 8)  = pa[2];
            *(float4*)(f + 12) = pa[3];
        }
        if (MODE == 1) {
            float g[16];
            const float4* ps = (const float4*)(ld_srow + kk);
            *(float4*)(g + 0)  = ps[0];
            *(float4*)(g + 4)  = ps[1];
            *(float4*)(g + 8)  = ps[2];
            *(float4*)(g + 12) = ps[3];
#pragma unroll
            for (int i = 0; i < 16; ++i) f[i] = g[i] - f[i];  // sum - state1
        }
        unsigned hi[8], lo[8];
#pragma unroll
        for (int j = 0; j < 8; ++j) {
            unsigned u0 = __float_as_uint(f[2 * j]);
            unsigned u1 = __float_as_uint(f[2 * j + 1]);
            float h0 = __uint_as_float(u0 & 0xFFFF0000u);
            float h1 = __uint_as_float(u1 & 0xFFFF0000u);
            hi[j] = pack_hi_pair(u0, u1);
            lo[j] = pack_hi_pair(__float_as_uint(f[2 * j] - h0),
                                 __float_as_uint(f[2 * j + 1] - h1));
        }
        ((uint4*)lda_hi)[0] = make_uint4(hi[0], hi[1], hi[2], hi[3]);
        ((uint4*)lda_hi)[1] = make_uint4(hi[4], hi[5], hi[6], hi[7]);
        ((uint4*)lda_lo)[0] = make_uint4(lo[0], lo[1], lo[2], lo[3]);
        ((uint4*)lda_lo)[1] = make_uint4(lo[4], lo[5], lo[6], lo[7]);

        if (kk + 32 < H_DIM) {
            __builtin_prefetch(ld_arow + kk + 32, 0, 0);
            if (MODE == 1) __builtin_prefetch(ld_srow + kk + 32, 0, 0);
        }

        wait_asynccnt0();  // this wave's DMA landed in LDS
        __syncthreads();   // everyone's stores + DMA visible

        // 3) fetch fragments from LDS (layouts match WMMA operand maps)
        Frag ah[2], al[2], bh[4], bl[4];
#pragma unroll
        for (int ms = 0; ms < 2; ++ms) {
            int rl = (wm * 32 + ms * 16 + l15) * LDS_STRIDE;
            ah[ms].q[0] = *(const uint4*)&lds_ahi[rl + 4 * half];
            ah[ms].q[1] = *(const uint4*)&lds_ahi[rl + 8 + 4 * half];
            al[ms].q[0] = *(const uint4*)&lds_alo[rl + 4 * half];
            al[ms].q[1] = *(const uint4*)&lds_alo[rl + 8 + 4 * half];
        }
#pragma unroll
        for (int ns = 0; ns < 4; ++ns) {
            int cl = (wn * 64 + ns * 16 + l15) * LDS_STRIDE + 4 * half;
            bh[ns].q[0] = *(const uint4*)&lds_bhi[cl];
            bh[ns].q[1] = *(const uint4*)&lds_bhi[cl + 8];
            bl[ns].q[0] = *(const uint4*)&lds_blo[cl];
            bl[ns].q[1] = *(const uint4*)&lds_blo[cl + 8];
        }

        // 4) bf16x3 WMMA: hi*hi + lo*hi + hi*lo, fp32 accumulate
#pragma unroll
        for (int ms = 0; ms < 2; ++ms)
#pragma unroll
            for (int ns = 0; ns < 4; ++ns) {
                acc[ms][ns] = __builtin_amdgcn_wmma_f32_16x16x32_bf16(
                    false, ah[ms].v, false, bh[ns].v, (short)0, acc[ms][ns], false, false);
                acc[ms][ns] = __builtin_amdgcn_wmma_f32_16x16x32_bf16(
                    false, al[ms].v, false, bh[ns].v, (short)0, acc[ms][ns], false, false);
                acc[ms][ns] = __builtin_amdgcn_wmma_f32_16x16x32_bf16(
                    false, ah[ms].v, false, bl[ns].v, (short)0, acc[ms][ns], false, false);
            }
    }

    // Epilogue: C/D layout -> VGPR r holds M = r + 8*half, N = l15.
#pragma unroll
    for (int ms = 0; ms < 2; ++ms)
#pragma unroll
        for (int ns = 0; ns < 4; ++ns) {
            int n = nbase + ns * 16 + l15;
            float bv = bias[s * H_DIM + n];
            int mrow = mbase + ms * 16 + half * 8;
            float* op = out + ((size_t)mrow * S_DIM + s) * H_DIM + n;
#pragma unroll
            for (int r = 0; r < 8; ++r)
                op[(size_t)r * S_DIM * H_DIM] = acc[ms][ns][r] + bv;
        }
}

extern "C" void kernel_launch(void* const* d_in, const int* in_sizes, int n_in,
                              void* d_out, int out_size, void* d_ws, size_t ws_size,
                              hipStream_t stream) {
    const float* x  = (const float*)d_in[0];
    const float* W1 = (const float*)d_in[1];
    const float* b1 = (const float*)d_in[2];
    const float* W3 = (const float*)d_in[3];
    const float* b3 = (const float*)d_in[4];
    float* out = (float*)d_out;

    // Workspace layout: state1 (fp32) | sums (fp32) | W1hi | W1lo | W3hi | W3lo
    const size_t kState1 = (size_t)B_DIM * S_DIM * H_DIM;
    const size_t kSum    = (size_t)B_DIM * H_DIM;
    const size_t kWpk    = (size_t)S_DIM * H_DIM * H_DIM / 2;  // u32 (2 bf16/word)
    float* state1 = (float*)d_ws;
    float* sums   = state1 + kState1;
    unsigned int* w1hi = (unsigned int*)(sums + kSum);
    unsigned int* w1lo = w1hi + kWpk;
    unsigned int* w3hi = w1lo + kWpk;
    unsigned int* w3lo = w3hi + kWpk;

    // 1) split both weight tensors into bf16 hi/lo.
    const int n2 = (int)kWpk;
    split_bf16x2_kernel<<<n2 / 256, 256, 0, stream>>>(W1, w1hi, w1lo, n2);
    split_bf16x2_kernel<<<n2 / 256, 256, 0, stream>>>(W3, w3hi, w3lo, n2);

    // 2) state1 = x @ W1^T + b1 (per sensor).
    dim3 grid(H_DIM / 128, B_DIM / 128, S_DIM);  // (4, 8, 64)
    sensor_gemm_kernel<0><<<grid, 256, 0, stream>>>(x, nullptr, w1hi, w1lo, b1, state1);

    // 3) sums[b,h] = sum_s state1.
    reduce_sum_kernel<<<(B_DIM * H_DIM) / 256, 256, 0, stream>>>(state1, sums);

    // 4) out = (sums - state1) @ W3^T + b3 (per sensor).
    sensor_gemm_kernel<1><<<grid, 256, 0, stream>>>(state1, sums, w3hi, w3lo, b3, out);
}
